// KnowledgeModel_86208583565456
// MI455X (gfx1250) — matrix operand order repaired and checked by reading, not compiled, not run on hardware
//
#include <hip/hip_runtime.h>

#define F_DIM 256
#define NREL 8

typedef __attribute__((ext_vector_type(16))) __bf16 v16bf;
typedef __attribute__((ext_vector_type(8)))  float  v8f;

union Frag16 {
    v16bf v;
    unsigned short u[16];
    unsigned int d[8];
};
static_assert(sizeof(Frag16) == 32, "frag size");

__device__ __forceinline__ unsigned short f2bf(float f) {
    unsigned int u = __builtin_bit_cast(unsigned int, f);
    u += 0x7FFFu + ((u >> 16) & 1u);          // round-to-nearest-even
    return (unsigned short)(u >> 16);
}

// ---- B fragment from pre-packed fragment-major layout:
// P dword index = ((tileks)*32 + lane)*8 + j   (32B contiguous per lane)
__device__ __forceinline__ void load_b_packed(Frag16& b, const unsigned int* P,
                                              int tileks, int lane) {
    const uint4* p = (const uint4*)(P + ((size_t)tileks * 32 + lane) * 8);
    uint4 lo = p[0], hi = p[1];
    b.d[0] = lo.x; b.d[1] = lo.y; b.d[2] = lo.z; b.d[3] = lo.w;
    b.d[4] = hi.x; b.d[5] = hi.y; b.d[6] = hi.z; b.d[7] = hi.w;
}

// ---- A fragment from LDS-staged, XOR-swizzled tile.
// Tile: 64 rows x (stride4) uint4 cols; element cols of row r live at col^(r&15).
// Fragment row for mtile t is t*16+ln; K pairs: lo col = ks*4+half, hi col = +2.
__device__ __forceinline__ void load_a_lds(Frag16& a, const uint4* As4, int stride4,
                                           int row, int ks, int half, int ln) {
    int cl = (ks * 4 + half) ^ ln;
    int ch = (ks * 4 + half + 2) ^ ln;
    uint4 lo = As4[row * stride4 + cl];
    uint4 hi = As4[row * stride4 + ch];
    a.d[0] = lo.x; a.d[1] = lo.y; a.d[2] = lo.z; a.d[3] = lo.w;
    a.d[4] = hi.x; a.d[5] = hi.y; a.d[6] = hi.z; a.d[7] = hi.w;
}

__device__ __forceinline__ v8f wmma_bf16(const Frag16& a, const Frag16& b, v8f c) {
    return __builtin_amdgcn_wmma_f32_16x16x32_bf16(
        false, a.v, false, b.v, (short)0, c, false, false);
}

// ---------------------------------------------------------------- elementwise
__global__ __launch_bounds__(256) void k_to_bf16(const float* in, unsigned short* out, int n) {
    int i = blockIdx.x * blockDim.x + threadIdx.x;
    if (i < n) out[i] = f2bf(in[i]);
}

__global__ __launch_bounds__(256) void k_relu_bf16(const float* in, unsigned short* out, int n) {
    int i = blockIdx.x * blockDim.x + threadIdx.x;
    if (i < n) out[i] = f2bf(fmaxf(in[i], 0.0f));
}

// ---- fused convert + pack of a K x 256 fp32 weight matrix (nmat of them) into
//      WMMA-B fragment-major layout: [mat][ntile][ks][lane][8 dwords]
__global__ __launch_bounds__(256) void k_pack_b(const float* src, unsigned int* dstp,
                                                int K, int nmat) {
    int idx = blockIdx.x * blockDim.x + threadIdx.x;
    int perMat = K * 128;                      // dwords per matrix
    int total = nmat * perMat;
    if (idx >= total) return;
    int mat = idx / perMat;
    int rem = idx - mat * perMat;
    int j    = rem & 7;
    int lane = (rem >> 3) & 31;
    int t    = rem >> 8;                       // t = ntile*KS + ks
    int KS = K >> 5;
    int ntile = t / KS;
    int ks    = t - ntile * KS;
    int half = lane >> 4;
    int n = ntile * 16 + (lane & 15);
    int kg = ks * 32 + 2 * j + (half << 3) + ((j >> 2) << 3);
    const float* S = src + (size_t)mat * K * F_DIM;
    unsigned int lo = f2bf(S[(size_t)kg * F_DIM + n]);
    unsigned int hi = f2bf(S[(size_t)(kg + 1) * F_DIM + n]);
    dstp[idx] = lo | (hi << 16);
}

// ------------------------------------------------------------- graph prep
__global__ __launch_bounds__(256) void k_count(const int* dst, const int* rel,
                                               int* cnt, int* relCnt, int E) {
    int e = blockIdx.x * blockDim.x + threadIdx.x;
    if (e < E) {
        atomicAdd(&cnt[dst[e] * NREL + rel[e]], 1);
        atomicAdd(&relCnt[rel[e]], 1);
    }
}

__global__ void k_prefix(const int* relCnt, int* pOff, int* relCur) {
    if (threadIdx.x == 0 && blockIdx.x == 0) {
        int off = 0;
        for (int r = 0; r < NREL; ++r) {
            pOff[r] = off;
            relCur[r] = off;
            off += ((relCnt[r] + 63) >> 6) << 6;   // pad bucket to multiple of 64
        }
        pOff[NREL] = off;
    }
}

__global__ __launch_bounds__(256) void k_norm(const int* dst, const int* rel,
                                              const int* cnt, float* norm, int E) {
    int e = blockIdx.x * blockDim.x + threadIdx.x;
    if (e < E) {
        float c = (float)cnt[dst[e] * NREL + rel[e]];
        norm[e] = 1.0f / fmaxf(c, 1.0f);
    }
}

__global__ __launch_bounds__(256) void k_scatter(const int* rel, int* relCur,
                                                 int* slots, int E) {
    int e = blockIdx.x * blockDim.x + threadIdx.x;
    if (e < E) {
        int p = atomicAdd(&relCur[rel[e]], 1);
        slots[p] = e;
    }
}

// ----------------- H = A(bf16)[M,256] @ Bpacked + bias
// one block per 64 rows; A tile staged in LDS once, 8 waves x 2 ntiles each
__global__ __launch_bounds__(256) void k_gemm_bias(const unsigned short* A,
                                                   const unsigned int* Bp,  // packed, KS=8
                                                   const float* bias, float* H, int M) {
    __shared__ uint4 As4[64 * 32];             // 32 KB
    int tid = threadIdx.x;
    int mbase = blockIdx.x * 64;
#pragma unroll
    for (int i = 0; i < 8; ++i) {              // stage 64 x 256 bf16
        int flat = tid + i * 256;              // 0..2047
        int row = flat >> 5, col = flat & 31;
        int gr = mbase + row;
        if (gr >= M) gr = 0;                   // clamped garbage row, store skipped below
        As4[row * 32 + (col ^ (row & 15))] =
            *(const uint4*)(A + (size_t)gr * F_DIM + col * 8);
    }
    __syncthreads();

    int wave = tid >> 5, lane = tid & 31;
    int half = lane >> 4, ln = lane & 15;
    v8f acc[2][4] = {{{}, {}, {}, {}}, {{}, {}, {}, {}}};
#pragma unroll
    for (int ks = 0; ks < 8; ++ks) {
        Frag16 af[4];
#pragma unroll
        for (int t = 0; t < 4; ++t) load_a_lds(af[t], As4, 32, t * 16 + ln, ks, half, ln);
#pragma unroll
        for (int u = 0; u < 2; ++u) {
            int nt = wave * 2 + u;
            Frag16 bf;
            load_b_packed(bf, Bp, nt * 8 + ks, lane);
#pragma unroll
            for (int t = 0; t < 4; ++t) acc[u][t] = wmma_bf16(af[t], bf, acc[u][t]);
        }
    }
#pragma unroll
    for (int u = 0; u < 2; ++u) {
        int ncol = (wave * 2 + u) * 16 + ln;
        float bv = bias[ncol];
#pragma unroll
        for (int t = 0; t < 4; ++t) {
            int tr = mbase + t * 16;
            if (tr < M) {                      // wave-uniform per t
#pragma unroll
                for (int v = 0; v < 8; ++v) {
                    H[(size_t)(tr + v + 8 * half) * F_DIM + ncol] = acc[u][t][v] + bv;
                }
            }
        }
    }
}

// -------- per-edge messages: msg = x[src] @ W_r, 64 edges per block (LDS tile),
//          scaled by 1/deg, atomically accumulated into H[dst]
__global__ __launch_bounds__(256) void k_edge_msg(const unsigned short* Xb,
                                                  const unsigned int* Wp8,  // [8] packed
                                                  const int* slots, const int* pOff,
                                                  const int* src, const int* dst,
                                                  const float* norm,
                                                  const unsigned short* zrow,
                                                  float* H) {
    __shared__ uint4 As4[64 * 32];             // 32 KB
    int tid = threadIdx.x;
    int g0 = blockIdx.x * 64;
#pragma unroll
    for (int i = 0; i < 8; ++i) {              // stage gathered 64 x 256 bf16
        int flat = tid + i * 256;
        int row = flat >> 5, col = flat & 31;
        int e = slots[g0 + row];
        const unsigned short* base = (e >= 0) ? Xb + (size_t)src[e] * F_DIM : zrow;
        As4[row * 32 + (col ^ (row & 15))] = *(const uint4*)(base + col * 8);
    }
    __syncthreads();

    int r = NREL - 1;
#pragma unroll
    for (int i = 0; i < NREL; ++i) {
        if (g0 < pOff[i + 1]) { r = i; break; }
    }
    const unsigned int* Bp = Wp8 + (size_t)r * (16 * 8) * 32 * 8;

    int wave = tid >> 5, lane = tid & 31;
    int half = lane >> 4, ln = lane & 15;
    v8f acc[2][4] = {{{}, {}, {}, {}}, {{}, {}, {}, {}}};
#pragma unroll
    for (int ks = 0; ks < 8; ++ks) {
        Frag16 af[4];
#pragma unroll
        for (int t = 0; t < 4; ++t) load_a_lds(af[t], As4, 32, t * 16 + ln, ks, half, ln);
#pragma unroll
        for (int u = 0; u < 2; ++u) {
            int nt = wave * 2 + u;
            Frag16 bf;
            load_b_packed(bf, Bp, nt * 8 + ks, lane);
#pragma unroll
            for (int t = 0; t < 4; ++t) acc[u][t] = wmma_bf16(af[t], bf, acc[u][t]);
        }
    }
#pragma unroll
    for (int u = 0; u < 2; ++u) {
        int ncol = (wave * 2 + u) * 16 + ln;
#pragma unroll
        for (int t = 0; t < 4; ++t) {
#pragma unroll
            for (int v = 0; v < 8; ++v) {
                int e = slots[g0 + t * 16 + v + 8 * half];
                if (e >= 0) {
                    atomicAdd(&H[(size_t)dst[e] * F_DIM + ncol], acc[u][t][v] * norm[e]);
                }
            }
        }
    }
}

// -------- out[e] = concat(h[src_e], h[dst_e]) @ Wp + bp  (K=512, 64 edges/block)
__global__ __launch_bounds__(256) void k_proj(const unsigned short* Hb,
                                              const unsigned int* Wpp,  // packed, KS=16
                                              const float* bp,
                                              const int* src, const int* dst,
                                              float* out) {
    __shared__ uint4 As4[64 * 64];             // 64 KB: 64 rows x 512 bf16
    int tid = threadIdx.x;
    int g0 = blockIdx.x * 64;
#pragma unroll
    for (int i = 0; i < 16; ++i) {             // stage concat(h[src], h[dst]) rows
        int flat = tid + i * 256;              // 0..4095
        int row = flat >> 6, col = flat & 63;
        int e = g0 + row;
        const unsigned short* base = (col < 32)
            ? Hb + (size_t)src[e] * F_DIM + col * 8
            : Hb + (size_t)dst[e] * F_DIM + (col - 32) * 8;
        As4[row * 64 + (col ^ (row & 15))] = *(const uint4*)base;
    }
    __syncthreads();

    int wave = tid >> 5, lane = tid & 31;
    int half = lane >> 4, ln = lane & 15;
    v8f acc[2][4] = {{{}, {}, {}, {}}, {{}, {}, {}, {}}};
#pragma unroll
    for (int ks = 0; ks < 16; ++ks) {
        Frag16 af[4];
#pragma unroll
        for (int t = 0; t < 4; ++t) load_a_lds(af[t], As4, 64, t * 16 + ln, ks, half, ln);
#pragma unroll
        for (int u = 0; u < 2; ++u) {
            int nt = wave * 2 + u;
            Frag16 bf;
            load_b_packed(bf, Wpp, nt * 16 + ks, lane);
#pragma unroll
            for (int t = 0; t < 4; ++t) acc[u][t] = wmma_bf16(af[t], bf, acc[u][t]);
        }
    }
#pragma unroll
    for (int u = 0; u < 2; ++u) {
        int ncol = (wave * 2 + u) * 16 + ln;
        float bv = bp[ncol];
#pragma unroll
        for (int t = 0; t < 4; ++t) {
#pragma unroll
            for (int v = 0; v < 8; ++v) {
                int row = g0 + t * 16 + v + 8 * half;
                out[(size_t)row * F_DIM + ncol] = acc[u][t][v] + bv;
            }
        }
    }
}

// --------------------------------------------------------------------- launch
extern "C" void kernel_launch(void* const* d_in, const int* in_sizes, int n_in,
                              void* d_out, int out_size, void* d_ws, size_t ws_size,
                              hipStream_t stream) {
    (void)n_in; (void)out_size;
    const float* x     = (const float*)d_in[0];
    const int*   eidx  = (const int*)d_in[1];
    const int*   etype = (const int*)d_in[2];
    const float* W1    = (const float*)d_in[3];
    const float* root1 = (const float*)d_in[4];
    const float* b1    = (const float*)d_in[5];
    const float* W2    = (const float*)d_in[6];
    const float* root2 = (const float*)d_in[7];
    const float* b2    = (const float*)d_in[8];
    const float* Wp    = (const float*)d_in[9];
    const float* bp    = (const float*)d_in[10];
    float* out = (float*)d_out;

    const int N = in_sizes[0] / F_DIM;       // 50000
    const int E = in_sizes[2];               // 400000
    const int* src = eidx;
    const int* dst = eidx + E;

    const int nquads_max = (E + 63) / 64 + NREL;   // worst-case 64-padded blocks
    const int nslots     = nquads_max * 64;

    // ---- workspace carve-up (256B aligned chunks)
    size_t off = 0;
    auto carve = [&](size_t bytes) -> void* {
        void* p = (char*)d_ws + off;
        off += (bytes + 255) & ~(size_t)255;
        return p;
    };
    unsigned short* x_bf  = (unsigned short*)carve((size_t)N * F_DIM * 2);
    unsigned short* h_bf  = (unsigned short*)carve((size_t)N * F_DIM * 2);
    float*          H     = (float*)        carve((size_t)N * F_DIM * 4);
    unsigned int*   W1p   = (unsigned int*) carve((size_t)NREL * F_DIM * 128 * 4);
    unsigned int*   W2p   = (unsigned int*) carve((size_t)NREL * F_DIM * 128 * 4);
    unsigned int*   r1p   = (unsigned int*) carve((size_t)F_DIM * 128 * 4);
    unsigned int*   r2p   = (unsigned int*) carve((size_t)F_DIM * 128 * 4);
    unsigned int*   Wpp   = (unsigned int*) carve((size_t)2 * F_DIM * 128 * 4);
    int*            cnt   = (int*)  carve((size_t)N * NREL * 4);
    float*          norm  = (float*)carve((size_t)E * 4);
    int*            slots = (int*)  carve((size_t)nslots * 4);
    unsigned short* zrow  = (unsigned short*)carve(F_DIM * 2);
    int*            relCnt = (int*) carve(64);
    int*            pOff   = (int*) carve(64);
    int*            relCur = (int*) carve(64);
    if (off > ws_size) return;               // workspace too small: bail deterministically

    auto cdiv = [](long long a, long long b) { return (int)((a + b - 1) / b); };
    const int NF = N * F_DIM;

    // ---- init
    hipMemsetAsync(cnt, 0, (size_t)N * NREL * 4, stream);
    hipMemsetAsync(relCnt, 0, 64, stream);
    hipMemsetAsync(slots, 0xFF, (size_t)nslots * 4, stream);   // -1 everywhere
    hipMemsetAsync(zrow, 0, F_DIM * 2, stream);

    // ---- converts + weight packing (fused fp32->bf16 + fragment-major layout)
    k_to_bf16<<<cdiv(NF, 256), 256, 0, stream>>>(x, x_bf, NF);
    k_pack_b<<<cdiv(NREL * F_DIM * 128, 256), 256, 0, stream>>>(W1, W1p, F_DIM, NREL);
    k_pack_b<<<cdiv(NREL * F_DIM * 128, 256), 256, 0, stream>>>(W2, W2p, F_DIM, NREL);
    k_pack_b<<<cdiv(F_DIM * 128, 256), 256, 0, stream>>>(root1, r1p, F_DIM, 1);
    k_pack_b<<<cdiv(F_DIM * 128, 256), 256, 0, stream>>>(root2, r2p, F_DIM, 1);
    k_pack_b<<<cdiv(2 * F_DIM * 128, 256), 256, 0, stream>>>(Wp, Wpp, 2 * F_DIM, 1);

    // ---- degree counts, norms, relation-bucketed (64-padded) edge ordering
    k_count  <<<cdiv(E, 256), 256, 0, stream>>>(dst, etype, cnt, relCnt, E);
    k_prefix <<<1, 32, 0, stream>>>(relCnt, pOff, relCur);
    k_norm   <<<cdiv(E, 256), 256, 0, stream>>>(dst, etype, cnt, norm, E);
    k_scatter<<<cdiv(E, 256), 256, 0, stream>>>(etype, relCur, slots, E);

    // ---- layer 1
    k_gemm_bias<<<cdiv(N, 64), 256, 0, stream>>>(x_bf, r1p, b1, H, N);
    k_edge_msg <<<nquads_max, 256, 0, stream>>>(x_bf, W1p, slots, pOff,
                                                src, dst, norm, zrow, H);
    k_relu_bf16<<<cdiv(NF, 256), 256, 0, stream>>>(H, h_bf, NF);

    // ---- layer 2 (h2 written into x_bf, dead after layer 1)
    k_gemm_bias<<<cdiv(N, 64), 256, 0, stream>>>(h_bf, r2p, b2, H, N);
    k_edge_msg <<<nquads_max, 256, 0, stream>>>(h_bf, W2p, slots, pOff,
                                                src, dst, norm, zrow, H);
    k_relu_bf16<<<cdiv(NF, 256), 256, 0, stream>>>(H, x_bf, NF);

    // ---- triplet projection
    k_proj<<<E / 64, 256, 0, stream>>>(x_bf, Wpp, bp, src, dst, out);
}
